// OutlierDeblur2_40870908788869
// MI455X (gfx1250) — compile-verified, matrix operand order checked by reading, not run on hardware
//
#include <hip/hip_runtime.h>
#include <cstddef>

// ---------------------------------------------------------------------------
// OutlierDeblur on MI455X (gfx1250, wave32):
//  - circular 33x33 PSF convolution as bf16 WMMA (Toeplitz 2-stage, f32 acc)
//    K = 32 WMMA taps (dx 0..31) + dx=32 tap as VALU rank-1 epilogue
//  - rows segmented 4x for occupancy; deterministic slab+combine
//  - halo rows fetched with global_load_async_to_lds_b64 (ASYNCcnt path)
//  - 3x3 regularizer stencils / CG vector ops in VALU
// ---------------------------------------------------------------------------

#define HP 544
#define WP 544
#define HS 512
#define WS 512
#define PS 16
#define NIMG 6
#define KK 33
#define KROW 34            // padded psf row stride (keeps (dx,dx+1) pairs aligned)
#define NSEG 4
#define SEGR (HP / NSEG)   // 136 rows processed per segment
#define SEGO (SEGR + 2*PS) // 168 output rows touched per segment
#define NTILE (WP / 16)    // 34 x-tiles
#define RBUF 128           // row buffer entries (32 lanes x 4 bf16, tail unused)

typedef __attribute__((ext_vector_type(16))) __bf16 v16bf;
typedef __attribute__((ext_vector_type(8)))  float  v8f;

static constexpr size_t N6  = (size_t)NIMG * HP * WP;   // padded image elems
static constexpr size_t NC6 = (size_t)NIMG * HS * WS;   // cropped image elems
static constexpr size_t SLABN = (size_t)NIMG * NTILE * NSEG * SEGO * 16;

__constant__ float c_GK[5][9] = {
  { 0.f, 0.f, 0.f,   0.f,-1.f, 1.f,   0.f, 0.f, 0.f},
  { 0.f, 0.f, 0.f,   0.f,-1.f, 0.f,   0.f, 1.f, 0.f},
  { 0.f, 0.f, 0.f,   1.f,-2.f, 1.f,   0.f, 0.f, 0.f},
  { 0.f, 1.f, 0.f,   0.f,-2.f, 0.f,   0.f, 1.f, 0.f},
  {-1.f, 1.f, 0.f,   1.f,-1.f, 0.f,   0.f, 0.f, 0.f}};
__constant__ float c_W08[5] = {0.1f, 0.1f, 0.025f, 0.025f, 0.025f};

__device__ __forceinline__ unsigned short f2bf(float f) {
  unsigned int u = __float_as_uint(f);
  u += 0x7FFFu + ((u >> 16) & 1u);          // round-to-nearest-even
  return (unsigned short)(u >> 16);
}
__device__ __forceinline__ __bf16 bfbits(unsigned short s) {
  __bf16 h; __builtin_memcpy(&h, &s, 2); return h;
}
__device__ __forceinline__ float bf2f(unsigned short s) {
  return __uint_as_float((unsigned)s << 16);   // exact bf16 -> f32
}

// --------------------------- PSF pack (f32 -> bf16, normal + flipped) -------
__global__ void k_prep_psf(const float* __restrict__ kern, unsigned short* __restrict__ pack) {
  int idx = blockIdx.x * blockDim.x + threadIdx.x;
  if (idx >= 2 * 2 * KK * KROW) return;
  int dir = idx / (2 * KK * KROW);
  int rem = idx % (2 * KK * KROW);
  int b   = rem / (KK * KROW);
  int rc  = rem % (KK * KROW);
  int dy = rc / KROW, dx = rc % KROW;
  float v = 0.f;
  if (dx < KK) {
    int sy = dir ? (KK - 1 - dy) : dy;
    int sx = dir ? (KK - 1 - dx) : dx;
    v = kern[(size_t)b * KK * KK + (size_t)sy * KK + sx];
  }
  pack[idx] = f2bf(v);
}

// --------------------------- f32 -> bf16 mirror (optional premultiply) ------
__global__ void k_tobf16(const float* __restrict__ src, const float* __restrict__ mul,
                         unsigned short* __restrict__ dst) {
  size_t i = (size_t)blockIdx.x * blockDim.x + threadIdx.x;
  if (i >= N6) return;
  float v = src[i];
  if (mul) v *= mul[i];
  dst[i] = f2bf(v);
}

// --------------------------- WMMA circular 33x33 conv (segmented) -----------
// out[y][x] = sum_{dy,dx} psf[dy][dx] * in[(y+dy-16)%HP][(x+dx-16)%WP]
// Stage1: tmp[r][x][dy] = sum_dx in[r][x+dx-16]*psf[dy][dx]
//   dx 0..31 via WMMA (M=x(16), K=32, N=dy -> 3 blocks of 16),
//   dx = 32 as VALU rank-1 epilogue (avoids a 97%-zero second K-chunk).
// Stage2 (LDS): strip[rl+32-dy][x] += tmp   (segment-local, never wraps)
// Each block: one wave, rows [seg*136, seg*136+136), private 168x16 slab;
// k_combine sums the <=2 overlapping strips per row (deterministic).
__global__ void k_conv_psf(const unsigned short* __restrict__ srcBf, // [6][HP][WP] bf16
                           float* __restrict__ slab,                 // [6][34][NSEG][SEGO][16]
                           const unsigned short* __restrict__ psf /* dir-selected [2][33][34] */) {
  extern __shared__ char smem[];
  float*          acc    = (float*)smem;                       // [SEGO][16]
  unsigned short* rowbuf = (unsigned short*)(acc + SEGO * 16); // [RBUF]
  unsigned short* kbuf   = rowbuf + RBUF;                      // [33*34]

  const int lane = threadIdx.x;            // one wave per workgroup
  const int n0 = lane & 15, hi = lane >> 4;
  const int tile = blockIdx.x;             // 0..33
  const int img  = blockIdx.y;             // 0..5  (b*3+c)
  const int seg  = blockIdx.z;             // 0..3
  const int b    = img / 3;
  const int x0   = tile * 16;

  const unsigned short* im = srcBf + (size_t)img * HP * WP;
  const unsigned short* kp = psf + (size_t)b * KK * KROW;
  float* slb = slab + ((((size_t)img * NTILE + tile) * NSEG) + seg) * (SEGO * 16);

  for (int q = lane; q < KK * KROW; q += 32) kbuf[q] = kp[q];
  for (int q = lane; q < SEGO * 16; q += 32) acc[q] = 0.f;
  __syncthreads();

  // B register tiles (K-chunk dx 0..31): B[k=dx][n=dy] = psf[dy][dx]
  v16bf Bm[3];
  float p32[3];                            // psf[dy][32], loop-invariant
  #pragma unroll
  for (int blk = 0; blk < 3; ++blk) {
    const int dy = n0 + 16 * blk;
    #pragma unroll
    for (int v = 0; v < 8; ++v) {
      const int kb = ((v < 4) ? 2 * v : 16 + 2 * (v - 4)) + hi * 8;
      unsigned short s0 = 0, s1 = 0;
      if (dy < KK) {
        s0 = kbuf[dy * KROW + kb];
        s1 = kbuf[dy * KROW + kb + 1];
      }
      Bm[blk][2 * v]     = bfbits(s0);
      Bm[blk][2 * v + 1] = bfbits(s1);
    }
    p32[blk] = (dy < KK) ? bf2f(kbuf[dy * KROW + 32]) : 0.f;
  }

  // async halo-row fetch: 32 lanes x 8B (4 bf16) = 256B per row (96 used +
  // harmless wrapped tail). Groups of 4 never split the wrap (544 % 4 == 0).
  const unsigned rowbufLds = (unsigned)(uintptr_t)rowbuf;  // flat low 32 = LDS offset
  const unsigned ldsAddr = rowbufLds + 8u * (unsigned)lane;
  int xs4 = x0 - PS + 4 * lane;
  xs4 += (xs4 < 0) ? WP : 0;  xs4 -= (xs4 >= WP) ? WP : 0;

  for (int rl = 0; rl < SEGR; ++rl) {
    const int r = seg * SEGR + rl;

    // WAR guard: prior ds reads of rowbuf must have sampled before overwrite
    asm volatile("s_wait_dscnt 0" ::: "memory");
    {
      unsigned gOff = (unsigned)((r * WP + xs4) * 2);
      asm volatile("global_load_async_to_lds_b64 %0, %1, %2"
                   :: "v"(ldsAddr), "v"(gOff), "s"(im) : "memory");
    }
    asm volatile("s_wait_asynccnt 0" ::: "memory");
    __syncthreads();

    // Toeplitz A: A[m][k] = row[x0 + m + k - 16]  (bf16 16x32 A-layout)
    v16bf A0;
    #pragma unroll
    for (int v = 0; v < 8; ++v) {
      const int kb = ((v < 4) ? 2 * v : 16 + 2 * (v - 4)) + hi * 8;
      A0[2 * v]     = bfbits(rowbuf[n0 + kb]);
      A0[2 * v + 1] = bfbits(rowbuf[n0 + kb + 1]);
    }

    v8f C0 = {}, C1 = {}, C2 = {};
    C0 = __builtin_amdgcn_wmma_f32_16x16x32_bf16(false, A0, false, Bm[0], (short)0, C0, false, false);
    C1 = __builtin_amdgcn_wmma_f32_16x16x32_bf16(false, A0, false, Bm[1], (short)0, C1, false, false);
    C2 = __builtin_amdgcn_wmma_f32_16x16x32_bf16(false, A0, false, Bm[2], (short)0, C2, false, false);

    // dx=32 rank-1 operand: row[m + 32] for this lane's 8 output columns
    float r32[8];
    #pragma unroll
    for (int j = 0; j < 8; ++j) r32[j] = bf2f(rowbuf[j + 8 * hi + 32]);

    // scatter: D[M=j+8*hi][N=n0] -> acc[rl + 32 - dy][x = j+8*hi]  (in [0,167])
    #pragma unroll
    for (int blk = 0; blk < 3; ++blk) {
      const int dy = n0 + 16 * blk;
      if (dy < KK) {
        const int rowl = rl + 32 - dy;
        const v8f& C = (blk == 0) ? C0 : (blk == 1) ? C1 : C2;
        #pragma unroll
        for (int j = 0; j < 8; ++j) {
          const int xl = j + 8 * hi;
          atomicAdd(&acc[rowl * 16 + xl], C[j] + p32[blk] * r32[j]);  // ds_add_f32
        }
      }
    }
    __syncthreads();
  }

  for (int q = lane; q < SEGO * 16; q += 32) slb[q] = acc[q];   // coalesced b32 stores
}

// combine segment slabs (<=2 overlap per row) with circular wrap; deterministic
__global__ void k_combine(const float* __restrict__ slab, float* __restrict__ dst) {
  size_t idx = (size_t)blockIdx.x * blockDim.x + threadIdx.x;
  if (idx >= N6) return;
  int x = (int)(idx % WP);
  int y = (int)((idx / WP) % HP);
  int img = (int)(idx / ((size_t)HP * WP));
  int tile = x >> 4, xl = x & 15;
  float v = 0.f;
  #pragma unroll
  for (int s = 0; s < NSEG; ++s) {
    int d = y - s * SEGR + PS;
    d += (d < 0) ? HP : 0;  d -= (d >= HP) ? HP : 0;
    if (d < SEGO)
      v += slab[((((size_t)img * NTILE + tile) * NSEG) + s) * (SEGO * 16) + d * 16 + xl];
  }
  dst[idx] = v;
}

// --------------------------- elementwise / stencil kernels ------------------
__global__ void k_pad_rep(const float* __restrict__ src, float* __restrict__ dst) {
  size_t idx = (size_t)blockIdx.x * blockDim.x + threadIdx.x;
  if (idx >= N6) return;
  int x = (int)(idx % WP);
  int y = (int)((idx / WP) % HP);
  int img = (int)(idx / ((size_t)HP * WP));
  int sy = y - PS; sy = sy < 0 ? 0 : (sy > HS - 1 ? HS - 1 : sy);
  int sx = x - PS; sx = sx < 0 ? 0 : (sx > WS - 1 ? WS - 1 : sx);
  dst[idx] = src[(size_t)img * HS * WS + (size_t)sy * WS + sx];
}

__global__ void k_pad_zero(const float* __restrict__ src, float* __restrict__ dst) {
  size_t idx = (size_t)blockIdx.x * blockDim.x + threadIdx.x;
  if (idx >= N6) return;
  int x = (int)(idx % WP);
  int y = (int)((idx / WP) % HP);
  int img = (int)(idx / ((size_t)HP * WP));
  float v = 0.f;
  if (y >= PS && y < PS + HS && x >= PS && x < PS + WS)
    v = src[(size_t)img * HS * WS + (size_t)(y - PS) * WS + (x - PS)];
  dst[idx] = v;
}

__global__ void k_ones_pad(float* __restrict__ dst) {
  size_t idx = (size_t)blockIdx.x * blockDim.x + threadIdx.x;
  if (idx >= N6) return;
  int x = (int)(idx % WP);
  int y = (int)((idx / WP) % HP);
  dst[idx] = (y >= PS && y < PS + HS && x >= PS && x < PS + WS) ? 1.f : 0.f;
}

// mode: 0 = corr3(x,g);  1 = w * corr3(x,g);  2 = L0.8 weight from corr3
__global__ void k_g(const float* __restrict__ src, float* __restrict__ dst,
                    const float* __restrict__ w, int gi, int mode) {
  size_t idx = (size_t)blockIdx.x * blockDim.x + threadIdx.x;
  if (idx >= N6) return;
  int x = (int)(idx % WP);
  int y = (int)((idx / WP) % HP);
  size_t base = idx - (size_t)y * WP - x;      // img*HP*WP
  float s = 0.f;
  #pragma unroll
  for (int a = 0; a < 3; ++a) {
    int yy = y + a - 1; if ((unsigned)yy >= (unsigned)HP) continue;
    #pragma unroll
    for (int bb = 0; bb < 3; ++bb) {
      int xx = x + bb - 1; if ((unsigned)xx >= (unsigned)WP) continue;
      s += c_GK[gi][a * 3 + bb] * src[base + (size_t)yy * WP + xx];
    }
  }
  if (mode == 1)      s *= w[idx];
  else if (mode == 2) s = c_W08[gi] * __powf(fmaxf(fabsf(s), 0.01f), -1.2f);
  dst[idx] = s;
}

// dst += LAMBD * conv3(src, g) (= corr with flipped g), zero-pad SAME
__global__ void k_gT_acc(const float* __restrict__ src, float* __restrict__ dst, int gi) {
  size_t idx = (size_t)blockIdx.x * blockDim.x + threadIdx.x;
  if (idx >= N6) return;
  int x = (int)(idx % WP);
  int y = (int)((idx / WP) % HP);
  size_t base = idx - (size_t)y * WP - x;
  float s = 0.f;
  #pragma unroll
  for (int a = 0; a < 3; ++a) {
    int yy = y + a - 1; if ((unsigned)yy >= (unsigned)HP) continue;
    #pragma unroll
    for (int bb = 0; bb < 3; ++bb) {
      int xx = x + bb - 1; if ((unsigned)xx >= (unsigned)WP) continue;
      s += c_GK[gi][(2 - a) * 3 + (2 - bb)] * src[base + (size_t)yy * WP + xx];
    }
  }
  dst[idx] += 0.003f * s;
}

__global__ void k_copy(float* __restrict__ d, const float* __restrict__ a) {
  size_t i = (size_t)blockIdx.x * blockDim.x + threadIdx.x;
  if (i < N6) d[i] = a[i];
}
__global__ void k_zero(float* __restrict__ d) {
  size_t i = (size_t)blockIdx.x * blockDim.x + threadIdx.x;
  if (i < N6) d[i] = 0.f;
}
__global__ void k_sub(float* __restrict__ r, const float* __restrict__ b, const float* __restrict__ q) {
  size_t i = (size_t)blockIdx.x * blockDim.x + threadIdx.x;
  if (i < N6) r[i] = b[i] - q[i];
}

__global__ void k_dot(const float* __restrict__ a, const float* __restrict__ b,
                      float* __restrict__ S, int off) {
  __shared__ float red[256];
  const int img = blockIdx.y;
  const size_t base = (size_t)img * HP * WP;
  float s = 0.f;
  for (size_t i = (size_t)blockIdx.x * blockDim.x + threadIdx.x;
       i < (size_t)HP * WP; i += (size_t)gridDim.x * blockDim.x)
    s += a[base + i] * b[base + i];
  red[threadIdx.x] = s;
  __syncthreads();
  for (int st = 128; st > 0; st >>= 1) {
    if ((int)threadIdx.x < st) red[threadIdx.x] += red[threadIdx.x + st];
    __syncthreads();
  }
  if (threadIdx.x == 0) atomicAdd(&S[off + img], red[0]);
}

__global__ void k_setS(float* __restrict__ S, int off, int n, float v) {
  if ((int)threadIdx.x < n) S[off + threadIdx.x] = v;
}
__global__ void k_copyS(float* __restrict__ S) {           // rho_1 = rho
  if (threadIdx.x < 6) S[6 + threadIdx.x] = S[threadIdx.x];
}
__global__ void k_updateP(float* __restrict__ p, const float* __restrict__ r,
                          const float* __restrict__ S) {
  size_t i = (size_t)blockIdx.x * blockDim.x + threadIdx.x;
  if (i >= N6) return;
  int img = (int)(i / ((size_t)HP * WP));
  float beta = S[img] / S[6 + img];
  p[i] = r[i] + beta * p[i];
}
__global__ void k_updateXR(float* __restrict__ x, float* __restrict__ r,
                           const float* __restrict__ p, const float* __restrict__ q,
                           const float* __restrict__ S) {
  size_t i = (size_t)blockIdx.x * blockDim.x + threadIdx.x;
  if (i >= N6) return;
  int img = (int)(i / ((size_t)HP * WP));
  float alp = S[img] / S[12 + img];
  x[i] += alp * p[i];
  r[i] -= alp * q[i];
}

__global__ void k_crop(const float* __restrict__ src, float* __restrict__ outs,
                       float* __restrict__ prev) {
  size_t idx = (size_t)blockIdx.x * blockDim.x + threadIdx.x;
  if (idx >= NC6) return;
  int x = (int)(idx % WS);
  int y = (int)((idx / WS) % HS);
  int img = (int)(idx / ((size_t)HS * WS));
  float v = src[(size_t)img * HP * WP + (size_t)(y + PS) * WP + (x + PS)];
  if (outs) outs[idx] = v;
  prev[idx] = v;
}

// --------------------------- host orchestration -----------------------------
extern "C" void kernel_launch(void* const* d_in, const int* in_sizes, int n_in,
                              void* d_out, int out_size, void* d_ws, size_t ws_size,
                              hipStream_t stream) {
  const float* blur = (const float*)d_in[0];   // [2,3,512,512]
  const float* kern = (const float*)d_in[1];   // [2,1,33,33]
  const float* mask = (const float*)d_in[2];   // [2,3,512,512]
  float* out = (float*)d_out;                  // [5,6,512,512]

  // workspace carve
  size_t off = 0;
  auto carve = [&](size_t bytes) -> void* {
    void* p = (char*)d_ws + off;
    off += (bytes + 255) & ~(size_t)255;
    return p;
  };
  const size_t imgB = N6 * sizeof(float);
  float* inp_p  = (float*)carve(imgB);
  float* mask_p = (float*)carve(imgB);
  float* ones_p = (float*)carve(imgB);
  float* bvec   = (float*)carve(imgB);
  float* Xv     = (float*)carve(imgB);
  float* Rv     = (float*)carve(imgB);
  float* Pv     = (float*)carve(imgB);
  float* Qv     = (float*)carve(imgB);
  float* T1     = (float*)carve(imgB);
  float* T2     = (float*)carve(imgB);
  float* Wb     = (float*)carve(5 * imgB);
  float* prev   = (float*)carve(NC6 * sizeof(float));
  float* slab   = (float*)carve(SLABN * sizeof(float));
  unsigned short* Vb  = (unsigned short*)carve(N6 * sizeof(unsigned short));
  unsigned short* T1b = (unsigned short*)carve(N6 * sizeof(unsigned short));
  unsigned short* psfPack = (unsigned short*)carve(2 * 2 * KK * KROW * sizeof(unsigned short));
  float* S      = (float*)carve(32 * sizeof(float));   // [0..5]=rho [6..11]=rho_1 [12..17]=sigma
  if (off > ws_size) return;                           // insufficient scratch: no-op

  const unsigned g6  = (unsigned)((N6  + 255) / 256);
  const unsigned gc6 = (unsigned)((NC6 + 255) / 256);
  const size_t convLds = (size_t)SEGO * 16 * sizeof(float)
                       + RBUF * sizeof(unsigned short)
                       + (size_t)KK * KROW * sizeof(unsigned short);

  auto CONV = [&](const unsigned short* sB, float* d, int dir) {
    k_conv_psf<<<dim3(NTILE, NIMG, NSEG), 32, convLds, stream>>>(
        sB, slab, psfPack + (size_t)dir * 2 * KK * KROW);
    k_combine<<<g6, 256, 0, stream>>>(slab, d);
  };
  // A(v) = convN(mask .* convF(v)) + LAMBD * sum_i g_i^T (w_i .* g_i v)
  auto APPLY_A = [&](const float* v, const float* msk, const float* wts) {
    k_tobf16<<<g6, 256, 0, stream>>>(v, nullptr, Vb);
    CONV(Vb, T1, /*dir=*/1);                            // flipped-kernel correlation
    k_tobf16<<<g6, 256, 0, stream>>>(T1, msk, T1b);     // fused *mask
    CONV(T1b, Qv, /*dir=*/0);                           // normal-kernel correlation
    for (int i = 0; i < 5; ++i) {
      k_g<<<g6, 256, 0, stream>>>(v, T2, wts ? wts + (size_t)i * N6 : nullptr,
                                  i, wts ? 1 : 0);
      k_gT_acc<<<g6, 256, 0, stream>>>(T2, Qv, i);
    }
  };
  auto CG = [&](const float* msk, const float* wts) {
    APPLY_A(Xv, msk, wts);
    k_sub<<<g6, 256, 0, stream>>>(Rv, bvec, Qv);      // r0 = b - A(x0)
    k_zero<<<g6, 256, 0, stream>>>(Pv);               // p0 = 0
    k_setS<<<1, 32, 0, stream>>>(S, 6, 6, 1.f);       // rho_1 = 1
    for (int it = 0; it < 15; ++it) {
      k_setS<<<1, 32, 0, stream>>>(S, 0, 6, 0.f);
      k_dot<<<dim3(64, NIMG), 256, 0, stream>>>(Rv, Rv, S, 0);      // rho
      k_updateP<<<g6, 256, 0, stream>>>(Pv, Rv, S);                 // p = r + (rho/rho_1)p
      APPLY_A(Pv, msk, wts);                                        // q = A(p)
      k_setS<<<1, 32, 0, stream>>>(S, 12, 6, 0.f);
      k_dot<<<dim3(64, NIMG), 256, 0, stream>>>(Pv, Qv, S, 12);     // <p,q>
      k_updateXR<<<g6, 256, 0, stream>>>(Xv, Rv, Pv, Qv, S);        // x += a p; r -= a q
      k_copyS<<<1, 32, 0, stream>>>(S);                             // rho_1 = rho
    }
  };

  // setup
  k_prep_psf<<<(2 * 2 * KK * KROW + 255) / 256, 256, 0, stream>>>(kern, psfPack);
  k_pad_rep<<<g6, 256, 0, stream>>>(blur, inp_p);
  k_pad_zero<<<g6, 256, 0, stream>>>(mask, mask_p);
  k_ones_pad<<<g6, 256, 0, stream>>>(ones_p);

  // initial L2 deconv (mask = padded ones, no weights)
  k_tobf16<<<g6, 256, 0, stream>>>(inp_p, ones_p, T1b);   // b = conv(inp .* ones_p)
  CONV(T1b, bvec, 0);
  k_copy<<<g6, 256, 0, stream>>>(Xv, inp_p);
  CG(ones_p, nullptr);
  k_crop<<<gc6, 256, 0, stream>>>(Xv, nullptr, prev);

  // 5 outer L0.8 deconvs
  for (int oi = 0; oi < 5; ++oi) {
    k_pad_rep<<<g6, 256, 0, stream>>>(prev, T1);                    // x0 padded
    for (int i = 0; i < 5; ++i)                                     // L0.8 weights
      k_g<<<g6, 256, 0, stream>>>(T1, Wb + (size_t)i * N6, nullptr, i, 2);
    k_copy<<<g6, 256, 0, stream>>>(Xv, T1);
    k_tobf16<<<g6, 256, 0, stream>>>(inp_p, mask_p, T1b);           // b = conv(inp .* mask)
    CONV(T1b, bvec, 0);
    CG(mask_p, Wb);
    k_crop<<<gc6, 256, 0, stream>>>(Xv, out + (size_t)oi * NC6, prev);
  }
}
